// VRWKV_Bottleneck_45028437131660
// MI455X (gfx1250) — matrix operand order repaired and checked by reading, not compiled, not run on hardware
//
#include <hip/hip_runtime.h>

typedef __attribute__((ext_vector_type(16))) __bf16 v16bf;
typedef __attribute__((ext_vector_type(8)))  float  v8f;

#define B_    32
#define HRES_ 25
#define WRES_ 25
#define T_    (HRES_ * WRES_)   // 625
#define C_    512
#define HID_  2048
#define M_    (B_ * T_)         // 20000
#define EPS_  1e-5f

enum { EP_STORE = 0, EP_SIGMOID = 1, EP_RESID = 2, EP_RELU2 = 3 };

// ---------------------------------------------------------------------------
// Pack an f32 weight W[K][N] (row-major) into per-lane WMMA B-fragment layout,
// split into bf16 hi + bf16 lo (hi = rne(x), lo = rne(x - hi)).
// Fragment for (ntile, ktile): 32 lanes x 64 bytes; lane holds 16 hi bf16 then
// 16 lo bf16, contiguous.  lane = (n%16) + 16*((k%32)/16), elem = k%16.
// ---------------------------------------------------------------------------
__global__ __launch_bounds__(256)
void pack_weight_k(const float* __restrict__ W, __bf16* __restrict__ out,
                   int K, int N) {
  size_t idx = (size_t)blockIdx.x * 256 + threadIdx.x;
  if (idx >= (size_t)K * N) return;
  int k = (int)(idx / N), n = (int)(idx % N);
  float f   = W[idx];
  __bf16 h  = (__bf16)f;
  __bf16 l  = (__bf16)(f - (float)h);
  int   KT  = K >> 5;
  int   lane = (n & 15) | (((k >> 4) & 1) << 4);
  size_t frag = ((size_t)(n >> 4) * KT + (k >> 5)) * 32 + lane;
  out[frag * 32 + (k & 15)]      = h;
  out[frag * 32 + 16 + (k & 15)] = l;
}

// ---------------------------------------------------------------------------
// bf16x2 split-precision WMMA GEMM:  C = epilogue(A[f32 MxK] @ Bpacked[KxN]).
// Block = 8 waves; block tile 32 x 512; wave tile 32 x 64 (2x4 WMMA tiles).
// A converted to bf16 hi/lo fragments on the fly; 3 wmma per tile per K-chunk.
// M must be a multiple of 32, N of 512, K of 32 (true for all calls here).
// ---------------------------------------------------------------------------
__global__ __launch_bounds__(256)
void gemm_wmma(const float* __restrict__ A, const v16bf* __restrict__ Bp,
               float* __restrict__ Cc, const float* __restrict__ base,
               const float* __restrict__ scale, int N, int K, int ep) {
  const int lane  = threadIdx.x & 31;
  const int wave  = threadIdx.x >> 5;
  const int lm    = lane & 15;
  const int lhalf = lane >> 4;
  const int m0 = blockIdx.x * 32;
  const int n0 = blockIdx.y * 512 + wave * 64;
  const int KT = K >> 5;

  v8f acc[2][4];
#pragma unroll
  for (int i = 0; i < 2; i++)
#pragma unroll
    for (int j = 0; j < 4; j++)
      acc[i][j] = (v8f){0.f, 0.f, 0.f, 0.f, 0.f, 0.f, 0.f, 0.f};

  // A fragment base: lane<16 covers k {0..15}, lane>=16 {16..31}
  const float* a0 = A + (size_t)(m0 + lm) * K + lhalf * 16;
  const float* a1 = a0 + (size_t)16 * K;

  for (int kc = 0; kc < KT; ++kc) {
    v16bf ah[2], al[2];
#pragma unroll
    for (int i = 0; i < 2; i++) {
      const float* p = (i == 0 ? a0 : a1) + kc * 32;
      float4 f0 = *(const float4*)(p);
      float4 f1 = *(const float4*)(p + 4);
      float4 f2 = *(const float4*)(p + 8);
      float4 f3 = *(const float4*)(p + 12);
      float vv[16] = {f0.x, f0.y, f0.z, f0.w, f1.x, f1.y, f1.z, f1.w,
                      f2.x, f2.y, f2.z, f2.w, f3.x, f3.y, f3.z, f3.w};
      v16bf h, l;
#pragma unroll
      for (int e = 0; e < 16; e++) {
        __bf16 hb = (__bf16)vv[e];
        h[e] = hb;
        l[e] = (__bf16)(vv[e] - (float)hb);
      }
      ah[i] = h;
      al[i] = l;
    }
    if (kc + 1 < KT) {  // hint next A chunk into cache (global_prefetch)
      __builtin_prefetch(a0 + (kc + 1) * 32, 0, 0);
      __builtin_prefetch(a1 + (kc + 1) * 32, 0, 0);
    }
    v16bf bh[4], bl[4];
#pragma unroll
    for (int j = 0; j < 4; j++) {
      size_t frag = ((size_t)((n0 >> 4) + j) * KT + kc) * 32 + lane;
      bh[j] = Bp[frag * 2];
      bl[j] = Bp[frag * 2 + 1];
    }
#pragma unroll
    for (int i = 0; i < 2; i++)
#pragma unroll
      for (int j = 0; j < 4; j++) {
        acc[i][j] = __builtin_amdgcn_wmma_f32_16x16x32_bf16(
            false, ah[i], false, bh[j], (short)0, acc[i][j], false, false);
        acc[i][j] = __builtin_amdgcn_wmma_f32_16x16x32_bf16(
            false, ah[i], false, bl[j], (short)0, acc[i][j], false, false);
        acc[i][j] = __builtin_amdgcn_wmma_f32_16x16x32_bf16(
            false, al[i], false, bh[j], (short)0, acc[i][j], false, false);
      }
  }

  // Epilogue.  C/D layout: elem r -> row m0+i*16+lhalf*8+r, col n0+j*16+lm.
#pragma unroll
  for (int i = 0; i < 2; i++)
#pragma unroll
    for (int j = 0; j < 4; j++)
#pragma unroll
      for (int r = 0; r < 8; r++) {
        int row = m0 + i * 16 + lhalf * 8 + r;
        int col = n0 + j * 16 + lm;
        size_t idx = (size_t)row * N + col;
        float v = acc[i][j][r];
        if (ep == EP_SIGMOID) {
          v = 1.f / (1.f + expf(-v));
        } else if (ep == EP_RELU2) {
          float t = fmaxf(v, 0.f);
          v = t * t;
        } else if (ep == EP_RESID) {
          v = base[idx] + (scale ? scale[idx] * v : v);
        }
        Cc[idx] = v;
      }
}

// ---------------------------------------------------------------------------
// Two-pass LayerNorm, one block (256 thr) per row; in-place safe.
// Optional post-multiply (used to fold sr into (sr*a) before @Wo).
// ---------------------------------------------------------------------------
__global__ __launch_bounds__(256)
void layernorm_k(const float* __restrict__ in, float* __restrict__ out,
                 const float* __restrict__ g, const float* __restrict__ b,
                 const float* __restrict__ mul, int rowlen) {
  __shared__ float red[256];
  const int row = blockIdx.x;
  const int nper = rowlen >> 8;  // 2 (C) or 8 (HID)
  const float* x = in + (size_t)row * rowlen;
  float* o = out + (size_t)row * rowlen;
  float vals[8];
  float s = 0.f;
  for (int i = 0; i < nper; i++) {
    vals[i] = x[threadIdx.x + (i << 8)];
    s += vals[i];
  }
  red[threadIdx.x] = s;
  __syncthreads();
  for (int st = 128; st > 0; st >>= 1) {
    if ((int)threadIdx.x < st) red[threadIdx.x] += red[threadIdx.x + st];
    __syncthreads();
  }
  float mean = red[0] / (float)rowlen;
  __syncthreads();
  float s2 = 0.f;
  for (int i = 0; i < nper; i++) {
    float d = vals[i] - mean;
    s2 += d * d;
  }
  red[threadIdx.x] = s2;
  __syncthreads();
  for (int st = 128; st > 0; st >>= 1) {
    if ((int)threadIdx.x < st) red[threadIdx.x] += red[threadIdx.x + st];
    __syncthreads();
  }
  float rs = rsqrtf(red[0] / (float)rowlen + EPS_);
  for (int i = 0; i < nper; i++) {
    int c = threadIdx.x + (i << 8);
    float v = (vals[i] - mean) * rs * g[c] + b[c];
    if (mul) v *= mul[(size_t)row * rowlen + c];
    o[c] = v;
  }
}

// ---------------------------------------------------------------------------
// q_shift (4 channel groups of 128, spatial shift +-1 in W / H, zero pad)
// fused with token mix: out = xn*mix + shift(xn)*(1-mix).
// ---------------------------------------------------------------------------
__global__ __launch_bounds__(256)
void qshift_mix_k(const float* __restrict__ xn, const float* __restrict__ mix,
                  float* __restrict__ out) {
  size_t idx = (size_t)blockIdx.x * 256 + threadIdx.x;
  if (idx >= (size_t)M_ * C_) return;
  int c = (int)(idx & (C_ - 1));
  int m = (int)(idx >> 9);
  int pos = m % T_;
  int h = pos / WRES_, w = pos % WRES_;
  float xv = xn[idx];
  float xx = 0.f;
  int grp = c >> 7;  // 128-channel groups
  int sm = m;
  bool ok;
  if (grp == 0)      { ok = (w >= 1);        sm = m - 1;     }
  else if (grp == 1) { ok = (w < WRES_ - 1); sm = m + 1;     }
  else if (grp == 2) { ok = (h >= 1);        sm = m - WRES_; }
  else               { ok = (h < HRES_ - 1); sm = m + WRES_; }
  if (ok) xx = xn[(size_t)sm * C_ + c];
  float t = mix[c];
  out[idx] = xv * t + xx * (1.f - t);
}

// ---------------------------------------------------------------------------
// WKV recurrence: one thread per (b, c) channel, sequential over T=625.
// Writes y over k in place (k[b,t,c] is dead after its read at step t).
// ---------------------------------------------------------------------------
__global__ __launch_bounds__(256)
void wkv_k(float* __restrict__ kv, const float* __restrict__ vv,
           const float* __restrict__ sd, const float* __restrict__ sf) {
  int tid = blockIdx.x * 256 + threadIdx.x;  // 0 .. B*C-1
  int c = tid & (C_ - 1);
  int b = tid >> 9;
  float w = sd[c] * (1.f / (float)T_);
  float u = sf[c] * (1.f / (float)T_);
  float aa = 0.f, bb = 0.f, pp = -1e38f;
  size_t idx = (size_t)b * T_ * C_ + c;
  for (int t = 0; t < T_; t++, idx += C_) {
    float kt = kv[idx], vt = vv[idx];
    float ww = u + kt;
    float p = fmaxf(pp, ww);
    float e1 = expf(pp - p), e2 = expf(ww - p);
    kv[idx] = (e1 * aa + e2 * vt) / (e1 * bb + e2);
    float ww2 = pp + w;
    float p2 = fmaxf(ww2, kt);
    e1 = expf(ww2 - p2);
    e2 = expf(kt - p2);
    aa = e1 * aa + e2 * vt;
    bb = e1 * bb + e2;
    pp = p2;
  }
}

// ---------------------------------------------------------------------------
extern "C" void kernel_launch(void* const* d_in, const int* in_sizes, int n_in,
                              void* d_out, int out_size, void* d_ws,
                              size_t ws_size, hipStream_t stream) {
  (void)in_sizes; (void)n_in; (void)out_size; (void)ws_size;
  const float* x   = (const float*)d_in[0];
  const float* g1  = (const float*)d_in[1];
  const float* b1  = (const float*)d_in[2];
  const float* g2  = (const float*)d_in[3];
  const float* b2  = (const float*)d_in[4];
  const float* sdc = (const float*)d_in[5];   // spatial_decay
  const float* sfr = (const float*)d_in[6];   // spatial_first
  const float* smk = (const float*)d_in[7];
  const float* smv = (const float*)d_in[8];
  const float* smr = (const float*)d_in[9];
  const float* Wk  = (const float*)d_in[10];
  const float* Wv  = (const float*)d_in[11];
  const float* Wr  = (const float*)d_in[12];
  const float* Wo  = (const float*)d_in[13];
  const float* kng = (const float*)d_in[14];
  const float* knb = (const float*)d_in[15];
  const float* cmk = (const float*)d_in[16];
  const float* cmr = (const float*)d_in[17];
  const float* Wkf = (const float*)d_in[18];  // (C, HID)
  const float* Wvf = (const float*)d_in[19];  // (HID, C)
  const float* Wrf = (const float*)d_in[20];
  const float* cng = (const float*)d_in[21];
  const float* cnb = (const float*)d_in[22];
  float* out = (float*)d_out;

  char* ws = (char*)d_ws;
  const size_t SZW1 = (size_t)C_ * C_ * 4;     // packed 512x512 weight
  const size_t SZWF = (size_t)C_ * HID_ * 4;   // packed 512x2048 weight
  const size_t SZMC = (size_t)M_ * C_ * 4;     // (M, C) f32
  const size_t SZMH = (size_t)M_ * HID_ * 4;   // (M, HID) f32
  size_t o = 0;
  __bf16* Wk_p  = (__bf16*)(ws + o); o += SZW1;
  __bf16* Wv_p  = (__bf16*)(ws + o); o += SZW1;
  __bf16* Wr_p  = (__bf16*)(ws + o); o += SZW1;
  __bf16* Wo_p  = (__bf16*)(ws + o); o += SZW1;
  __bf16* Wkf_p = (__bf16*)(ws + o); o += SZWF;
  __bf16* Wvf_p = (__bf16*)(ws + o); o += SZWF;
  __bf16* Wrf_p = (__bf16*)(ws + o); o += SZW1;
  float* A1 = (float*)(ws + o); o += SZMC;  // xn, later hn
  float* A2 = (float*)(ws + o); o += SZMC;  // mix scratch (xk/xv/xr/xk2/xr2)
  float* A5 = (float*)(ws + o); o += SZMC;  // k -> a -> sr*LN(a)
  float* A6 = (float*)(ws + o); o += SZMC;  // v, later r2
  float* A7 = (float*)(ws + o); o += SZMC;  // sr
  float* A8 = (float*)(ws + o); o += SZMC;  // x1 residual
  float* A9 = (float*)(ws + o); o += SZMH;  // kk / kkn

  dim3 blk(256);
  const int g1sz = (C_ * C_) / 256;       // 1024 blocks
  const int gfsz = (C_ * HID_) / 256;     // 4096 blocks
  const int gmix = (M_ * C_) / 256;       // 40000 blocks
  const dim3 gemC(M_ / 32, 1);            // N = 512
  const dim3 gemH(M_ / 32, HID_ / 512);   // N = 2048

  // Pack all weights into bf16 hi/lo WMMA fragments.
  pack_weight_k<<<g1sz, blk, 0, stream>>>(Wk, Wk_p, C_, C_);
  pack_weight_k<<<g1sz, blk, 0, stream>>>(Wv, Wv_p, C_, C_);
  pack_weight_k<<<g1sz, blk, 0, stream>>>(Wr, Wr_p, C_, C_);
  pack_weight_k<<<g1sz, blk, 0, stream>>>(Wo, Wo_p, C_, C_);
  pack_weight_k<<<gfsz, blk, 0, stream>>>(Wkf, Wkf_p, C_, HID_);
  pack_weight_k<<<gfsz, blk, 0, stream>>>(Wvf, Wvf_p, HID_, C_);
  pack_weight_k<<<g1sz, blk, 0, stream>>>(Wrf, Wrf_p, C_, C_);

  // --- spatial-mix branch ---
  layernorm_k<<<M_, blk, 0, stream>>>(x, A1, g1, b1, nullptr, C_);   // xn
  qshift_mix_k<<<gmix, blk, 0, stream>>>(A1, smk, A2);               // xk
  gemm_wmma<<<gemC, blk, 0, stream>>>(A2, (const v16bf*)Wk_p, A5,
                                      nullptr, nullptr, C_, C_, EP_STORE);
  qshift_mix_k<<<gmix, blk, 0, stream>>>(A1, smv, A2);               // xv
  gemm_wmma<<<gemC, blk, 0, stream>>>(A2, (const v16bf*)Wv_p, A6,
                                      nullptr, nullptr, C_, C_, EP_STORE);
  qshift_mix_k<<<gmix, blk, 0, stream>>>(A1, smr, A2);               // xr
  gemm_wmma<<<gemC, blk, 0, stream>>>(A2, (const v16bf*)Wr_p, A7,
                                      nullptr, nullptr, C_, C_, EP_SIGMOID);
  wkv_k<<<(B_ * C_) / 256, blk, 0, stream>>>(A5, A6, sdc, sfr);      // a (in place)
  // (sr * LN(a)):  Python `sr * a @ Wo` == (sr*a) @ Wo
  layernorm_k<<<M_, blk, 0, stream>>>(A5, A5, kng, knb, A7, C_);
  gemm_wmma<<<gemC, blk, 0, stream>>>(A5, (const v16bf*)Wo_p, A8,
                                      x, nullptr, C_, C_, EP_RESID);  // x1

  // --- channel-mix (FFN) branch ---
  layernorm_k<<<M_, blk, 0, stream>>>(A8, A1, g2, b2, nullptr, C_);  // hn
  qshift_mix_k<<<gmix, blk, 0, stream>>>(A1, cmk, A2);               // xk2
  gemm_wmma<<<gemH, blk, 0, stream>>>(A2, (const v16bf*)Wkf_p, A9,
                                      nullptr, nullptr, HID_, C_, EP_RELU2);
  layernorm_k<<<M_, blk, 0, stream>>>(A9, A9, cng, cnb, nullptr, HID_);
  qshift_mix_k<<<gmix, blk, 0, stream>>>(A1, cmr, A2);               // xr2
  gemm_wmma<<<gemC, blk, 0, stream>>>(A2, (const v16bf*)Wrf_p, A6,
                                      nullptr, nullptr, C_, C_, EP_SIGMOID);
  gemm_wmma<<<gemC, blk, 0, stream>>>(A9, (const v16bf*)Wvf_p, out,
                                      A8, A6, C_, HID_, EP_RESID);
}
